// AttentionMambaBlock_1477468750545
// MI455X (gfx1250) — matrix operand order repaired and checked by reading, not compiled, not run on hardware
//
#include <hip/hip_runtime.h>
#include <hip/hip_bf16.h>

// ---------------------------------------------------------------------------
// Problem constants (match reference)
// ---------------------------------------------------------------------------
#define BB      2
#define DIM     192
#define HH      48
#define WW      48
#define HEADS   6
#define KWIN    3
#define DEPTH   12
#define D_STATE 16
#define D_CONV  4
#define D_INNER 384          // EXPAND * DIM
#define DT_RANK 12           // (DIM+15)/16
#define LTOK    (HH*WW)      // 2304 tokens per batch
#define NROW    (BB*LTOK)    // 4608 rows (multiple of 64)
#define DBC_N   (DT_RANK + 2*D_STATE)   // 44
#define DBC_LD  64                      // padded row stride for dbc (mult of 64)
#define DT_KP   32                      // dt_proj K padded 12 -> 32
#define EPSV    1e-5f

typedef __attribute__((ext_vector_type(16))) _Float16 v16h;
typedef __attribute__((ext_vector_type(8)))  _Float16 v8h;
typedef __attribute__((ext_vector_type(8)))  float    v8f;

__device__ __forceinline__ float silu_f(float x)     { return x / (1.0f + __expf(-x)); }
__device__ __forceinline__ float softplus_f(float x) { return (x > 20.0f) ? x : log1pf(__expf(x)); }

// ---------------------------------------------------------------------------
// WMMA GEMM:  O[M,Npad] = act( A16[M,Kpad] @ W16[Npad,Kpad]^T + bias )
//   - all operands pre-packed fp16, K padded to mult-of-32, N padded to
//     mult-of-64, M mult-of-64  ->  NO guards anywhere, pure b128 loads.
//   - block = 128 threads = 4 waves; wave w owns M-tile (blockIdx.y*64+w*16)
//     and the 4 N-tiles of column group blockIdx.x*64. A fragment is loaded
//     once per K-step and reused for 4 WMMAs.
// Fragment layouts (as used here, contiguous k, wave32):
//   A: m = lane&15 ; halves [k0+16*half, +16)
//   B: n = lane&15 ; halves [k0+16*half, +16)
//   D: n = lane&15 ; m = v + 8*half
// ---------------------------------------------------------------------------
template<int ACT, bool S32, bool S16>
__global__ void wmma_gemm_kernel(const _Float16* __restrict__ A16, int lda,
                                 const _Float16* __restrict__ W16, int ldw,
                                 const float*    __restrict__ bias,
                                 float*          __restrict__ O32,
                                 _Float16*       __restrict__ O16,
                                 int ldo, int Kpad)
{
    const int lane  = threadIdx.x & 31;
    const int wave  = threadIdx.x >> 5;
    const int half  = lane >> 4;
    const int mr    = lane & 15;

    const int tileM  = blockIdx.y * 64 + wave * 16;
    const int tileN0 = blockIdx.x * 64;

    const _Float16* arow = A16 + (size_t)(tileM + mr) * lda;

    v8f acc0 = {}, acc1 = {}, acc2 = {}, acc3 = {};

    for (int k0 = 0; k0 < Kpad; k0 += 32) {
        // A fragment: two contiguous 16B runs
        v8h a0 = *(const v8h*)(arow + k0 + 16 * half);
        v8h a1 = *(const v8h*)(arow + k0 + 16 * half + 8);
        v16h af = __builtin_shufflevector(a0, a1,
                    0,1,2,3,4,5,6,7,8,9,10,11,12,13,14,15);

#pragma unroll
        for (int tt = 0; tt < 4; ++tt) {
            const _Float16* wp = W16 + (size_t)(tileN0 + tt * 16 + mr) * ldw
                                 + k0 + 16 * half;
            v8h b0 = *(const v8h*)(wp);
            v8h b1 = *(const v8h*)(wp + 8);
            v16h bf = __builtin_shufflevector(b0, b1,
                        0,1,2,3,4,5,6,7,8,9,10,11,12,13,14,15);
            v8f c = (tt == 0) ? acc0 : (tt == 1) ? acc1 : (tt == 2) ? acc2 : acc3;
            c = __builtin_amdgcn_wmma_f32_16x16x32_f16(
                    false, af, false, bf, (short)0, c, false, false);
            if      (tt == 0) acc0 = c;
            else if (tt == 1) acc1 = c;
            else if (tt == 2) acc2 = c;
            else              acc3 = c;
        }
    }

#pragma unroll
    for (int tt = 0; tt < 4; ++tt) {
        v8f c = (tt == 0) ? acc0 : (tt == 1) ? acc1 : (tt == 2) ? acc2 : acc3;
        int n = tileN0 + tt * 16 + mr;
        float bv = (bias != nullptr) ? bias[n] : 0.0f;
#pragma unroll
        for (int v = 0; v < 8; ++v) {
            int m = tileM + v + (half << 3);
            float val = c[v] + bv;
            if (ACT == 1) val = softplus_f(val);
            if (S32) O32[(size_t)m * ldo + n] = val;
            if (S16) O16[(size_t)m * ldo + n] = (_Float16)val;
        }
    }
}

template<int ACT, bool S32, bool S16>
static inline void launch_gemm(const _Float16* A16, int lda,
                               const _Float16* W16, int ldw,
                               const float* bias,
                               float* O32, _Float16* O16,
                               int ldo, int M, int Npad, int Kpad,
                               hipStream_t stream)
{
    dim3 grid(Npad / 64, M / 64);
    dim3 block(128);
    wmma_gemm_kernel<ACT, S32, S16><<<grid, block, 0, stream>>>(
        A16, lda, W16, ldw, bias, O32, O16, ldo, Kpad);
}

// ---------------------------------------------------------------------------
// Weight pack: src fp32 [L, N, K] -> dst fp16 [L, Npad, Kpad], zero padded.
// ---------------------------------------------------------------------------
__global__ void pack_w_kernel(const float* __restrict__ src, _Float16* __restrict__ dst,
                              int L, int N, int K, int Npad, int Kpad)
{
    int idx = blockIdx.x * blockDim.x + threadIdx.x;
    int total = L * Npad * Kpad;
    if (idx >= total) return;
    int k = idx % Kpad;
    int n = (idx / Kpad) % Npad;
    int l = idx / (Kpad * Npad);
    float v = (n < N && k < K) ? src[((size_t)l * N + n) * K + k] : 0.0f;
    dst[idx] = (_Float16)v;
}

// ---------------------------------------------------------------------------
// NCHW fp32 -> (B, L, C) fp16 tokens
// ---------------------------------------------------------------------------
__global__ void reshape_in_kernel(const float* __restrict__ x, _Float16* __restrict__ t16)
{
    int idx = blockIdx.x * blockDim.x + threadIdx.x;   // over B*C*L
    if (idx >= BB * DIM * LTOK) return;
    int hw = idx % LTOK;
    int c  = (idx / LTOK) % DIM;
    int b  = idx / (LTOK * DIM);
    t16[((size_t)b * LTOK + hw) * DIM + c] = (_Float16)x[((size_t)b * DIM + c) * LTOK + hw];
}

// ---------------------------------------------------------------------------
// Depthwise causal conv (D_CONV=4) + SiLU; xin = xz[..., 0:384] fp32.
// Writes xc16 (fp16) for the x_proj GEMM / scan.
// ---------------------------------------------------------------------------
__global__ void conv_silu_kernel(const float* __restrict__ xz,
                                 const float* __restrict__ conv_w,
                                 const float* __restrict__ conv_b,
                                 _Float16* __restrict__ xc16, int layer)
{
    int idx = blockIdx.x * blockDim.x + threadIdx.x;   // over B*L*D_INNER
    if (idx >= NROW * D_INNER) return;
    int d   = idx % D_INNER;
    int row = idx / D_INNER;
    int l   = row % LTOK;
    int b   = row / LTOK;
    const float* w = conv_w + ((size_t)layer * D_INNER + d) * D_CONV;
    float acc = conv_b[layer * D_INNER + d];
#pragma unroll
    for (int j = 0; j < D_CONV; ++j) {
        int ll = l - (D_CONV - 1) + j;
        if (ll >= 0) acc += w[j] * xz[((size_t)(b * LTOK + ll)) * (2 * D_INNER) + d];
    }
    xc16[(size_t)row * D_INNER + d] = (_Float16)silu_f(acc);
}

// ---------------------------------------------------------------------------
// Selective scan: one thread per (b, d), 16 states in registers, fp32 math,
// hardware __expf per step.  Fuses +D*xc and *silu(z); writes u16 (fp16).
// ---------------------------------------------------------------------------
__global__ void scan_kernel(const float*    __restrict__ dt,
                            const float*    __restrict__ dbc,   // ld = DBC_LD
                            const _Float16* __restrict__ xc16,
                            const float*    __restrict__ xz,    // z at col 384
                            const float*    __restrict__ A_log,
                            const float*    __restrict__ D_skip,
                            _Float16*       __restrict__ u16, int layer)
{
    int idx = blockIdx.x * blockDim.x + threadIdx.x;   // over B*D_INNER = 768
    if (idx >= BB * D_INNER) return;
    int d = idx % D_INNER;
    int b = idx / D_INNER;

    float Areg[D_STATE], h[D_STATE];
#pragma unroll
    for (int s = 0; s < D_STATE; ++s) {
        Areg[s] = -__expf(A_log[((size_t)layer * D_INNER + d) * D_STATE + s]);
        h[s] = 0.0f;
    }
    const float Dd = D_skip[layer * D_INNER + d];

    for (int l = 0; l < LTOK; ++l) {
        size_t row = (size_t)b * LTOK + l;
        float dtv = dt[row * D_INNER + d];
        float xv  = (float)xc16[row * D_INNER + d];
        float zv  = xz[row * (2 * D_INNER) + D_INNER + d];
        const float* bc = dbc + row * DBC_LD + DT_RANK;   // B [0:16), C [16:32)
        float y  = 0.0f;
        float dx = dtv * xv;
#pragma unroll
        for (int s = 0; s < D_STATE; ++s) {
            h[s] = h[s] * __expf(dtv * Areg[s]) + dx * bc[s];
            y += h[s] * bc[D_STATE + s];
        }
        u16[row * D_INNER + d] = (_Float16)((y + Dd * xv) * silu_f(zv));
    }
}

// ---------------------------------------------------------------------------
// RMSNorm over C=192: read fp32 t, write fp16 t16 (input of qkv GEMM).
// ---------------------------------------------------------------------------
__global__ void rmsnorm_kernel(const float* __restrict__ t, const float* __restrict__ w,
                               _Float16* __restrict__ t16)
{
    int r = blockIdx.x * blockDim.x + threadIdx.x;
    if (r >= NROW) return;
    const float* p = t + (size_t)r * DIM;
    float s = 0.0f;
    for (int c = 0; c < DIM; ++c) s += p[c] * p[c];
    float inv = rsqrtf(s / (float)DIM + EPSV);
    _Float16* q = t16 + (size_t)r * DIM;
    for (int c = 0; c < DIM; ++c) q[c] = (_Float16)(p[c] * inv * w[c]);
}

// ---------------------------------------------------------------------------
// 3x3 neighborhood attention, one thread per (b,i,j,head); dh = 32.
// qkv row layout: [q(192) | k(192) | v(192)] fp32; writes attn16 fp16.
// ---------------------------------------------------------------------------
__global__ void na2d_kernel(const float* __restrict__ qkv,
                            const float* __restrict__ rpb,
                            _Float16* __restrict__ out16)
{
    int idx = blockIdx.x * blockDim.x + threadIdx.x;   // B*H*W*HEADS
    if (idx >= BB * LTOK * HEADS) return;
    int h   = idx % HEADS;
    int pix = idx / HEADS;
    int j   = pix % WW;
    int i   = (pix / WW) % HH;
    int b   = pix / LTOK;
    const int dh = DIM / HEADS;                        // 32
    const float scale = 0.17677669529663687f;          // 1/sqrt(32)

    size_t row = (size_t)b * LTOK + i * WW + j;
    const float* q = qkv + row * (3 * DIM) + h * dh;

    int ni = i - 1; if (ni < 0) ni = 0; if (ni > HH - KWIN) ni = HH - KWIN;
    int nj = j - 1; if (nj < 0) nj = 0; if (nj > WW - KWIN) nj = WW - KWIN;

    float logits[KWIN * KWIN];
    float mx = -1e30f;
#pragma unroll
    for (int a = 0; a < KWIN; ++a) {
#pragma unroll
        for (int c = 0; c < KWIN; ++c) {
            int ii = ni + a, jj = nj + c;
            const float* kk = qkv + ((size_t)b * LTOK + ii * WW + jj) * (3 * DIM) + DIM + h * dh;
            float dot = 0.0f;
            for (int e = 0; e < dh; ++e) dot += q[e] * kk[e];
            int ri = i - ii + (KWIN - 1);
            int rj = j - jj + (KWIN - 1);
            float lg = dot * scale + rpb[((size_t)h * (2*KWIN-1) + ri) * (2*KWIN-1) + rj];
            logits[a * KWIN + c] = lg;
            mx = fmaxf(mx, lg);
        }
    }
    float den = 0.0f;
#pragma unroll
    for (int n = 0; n < KWIN * KWIN; ++n) { logits[n] = __expf(logits[n] - mx); den += logits[n]; }
    float inv = 1.0f / den;

    float o[DIM / HEADS];
    for (int e = 0; e < dh; ++e) o[e] = 0.0f;
#pragma unroll
    for (int a = 0; a < KWIN; ++a) {
#pragma unroll
        for (int c = 0; c < KWIN; ++c) {
            int ii = ni + a, jj = nj + c;
            const float* vv = qkv + ((size_t)b * LTOK + ii * WW + jj) * (3 * DIM) + 2 * DIM + h * dh;
            float p = logits[a * KWIN + c] * inv;
            for (int e = 0; e < dh; ++e) o[e] += p * vv[e];
        }
    }
    _Float16* op = out16 + row * DIM + h * dh;
    for (int e = 0; e < dh; ++e) op[e] = (_Float16)o[e];
}

// ---------------------------------------------------------------------------
// (B, L, C) fp32 tokens -> NCHW fp32 output
// ---------------------------------------------------------------------------
__global__ void reshape_out_kernel(const float* __restrict__ t, float* __restrict__ y)
{
    int idx = blockIdx.x * blockDim.x + threadIdx.x;   // over B*C*L
    if (idx >= BB * DIM * LTOK) return;
    int hw = idx % LTOK;
    int c  = (idx / LTOK) % DIM;
    int b  = idx / (LTOK * DIM);
    y[((size_t)b * DIM + c) * LTOK + hw] = t[((size_t)b * LTOK + hw) * DIM + c];
}

// ---------------------------------------------------------------------------
// Host orchestration
// ---------------------------------------------------------------------------
extern "C" void kernel_launch(void* const* d_in, const int* in_sizes, int n_in,
                              void* d_out, int out_size, void* d_ws, size_t ws_size,
                              hipStream_t stream)
{
    (void)in_sizes; (void)n_in; (void)out_size; (void)ws_size;

    const float* x         = (const float*)d_in[0];
    const float* in_proj_w = (const float*)d_in[1];   // (12, 768, 192)
    const float* conv_w    = (const float*)d_in[2];   // (12, 384, 4)
    const float* conv_b    = (const float*)d_in[3];   // (12, 384)
    const float* x_proj_w  = (const float*)d_in[4];   // (12, 44, 384)
    const float* dt_proj_w = (const float*)d_in[5];   // (12, 384, 12)
    const float* dt_proj_b = (const float*)d_in[6];   // (12, 384)
    const float* A_log     = (const float*)d_in[7];   // (12, 384, 16)
    const float* D_skip    = (const float*)d_in[8];   // (12, 384)
    const float* out_proj_w= (const float*)d_in[9];   // (12, 192, 384)
    const float* norm_w    = (const float*)d_in[10];  // (192)
    const float* qkv_w     = (const float*)d_in[11];  // (576, 192)
    const float* qkv_b     = (const float*)d_in[12];  // (576)
    const float* rpb       = (const float*)d_in[13];  // (6, 5, 5)
    const float* proj_w    = (const float*)d_in[14];  // (192, 192)
    const float* proj_b    = (const float*)d_in[15];  // (192)
    float* out = (float*)d_out;

    // ---------------- workspace layout (256B aligned slices) ----------------
    size_t off = 0;
    char* base = (char*)d_ws;
    auto alloc_f32 = [&](size_t n) { float* p = (float*)(base + off);
                                     off += ((n * 4 + 255) / 256) * 256; return p; };
    auto alloc_f16 = [&](size_t n) { _Float16* p = (_Float16*)(base + off);
                                     off += ((n * 2 + 255) / 256) * 256; return p; };

    float*    xz     = alloc_f32((size_t)NROW * 2 * D_INNER);  // 4608x768
    float*    dbc32  = alloc_f32((size_t)NROW * DBC_LD);       // 4608x64
    float*    dt32   = alloc_f32((size_t)NROW * D_INNER);      // 4608x384
    float*    t32    = alloc_f32((size_t)NROW * DIM);          // 4608x192
    _Float16* t16    = alloc_f16((size_t)NROW * DIM);
    _Float16* xc16   = alloc_f16((size_t)NROW * D_INNER);
    _Float16* dbc16  = alloc_f16((size_t)NROW * DBC_LD);
    _Float16* u16    = alloc_f16((size_t)NROW * D_INNER);
    _Float16* w_in   = alloc_f16((size_t)DEPTH * 2 * D_INNER * DIM);   // 12x768x192
    _Float16* w_x    = alloc_f16((size_t)DEPTH * DBC_LD * D_INNER);    // 12x64x384
    _Float16* w_dt   = alloc_f16((size_t)DEPTH * D_INNER * DT_KP);     // 12x384x32
    _Float16* w_out  = alloc_f16((size_t)DEPTH * DIM * D_INNER);       // 12x192x384
    _Float16* w_qkv  = alloc_f16((size_t)3 * DIM * DIM);               // 576x192
    _Float16* w_proj = alloc_f16((size_t)DIM * DIM);                   // 192x192

    float*    qkvb   = xz;     // reuse: 4608x576 <= 4608x768
    float*    tn     = dt32;   // reuse: 4608x192 <= 4608x384
    _Float16* attn16 = xc16;   // reuse: 4608x192 <= 4608x384

    const int TB = 256;
    auto nblk = [](int n, int tb) { return (n + tb - 1) / tb; };

    // ---------------- pack weights to fp16 (padded) ----------------
    {
        int n;
        n = DEPTH * 2 * D_INNER * DIM;
        pack_w_kernel<<<nblk(n, TB), TB, 0, stream>>>(in_proj_w, w_in,
            DEPTH, 2 * D_INNER, DIM, 2 * D_INNER, DIM);
        n = DEPTH * DBC_LD * D_INNER;
        pack_w_kernel<<<nblk(n, TB), TB, 0, stream>>>(x_proj_w, w_x,
            DEPTH, DBC_N, D_INNER, DBC_LD, D_INNER);
        n = DEPTH * D_INNER * DT_KP;
        pack_w_kernel<<<nblk(n, TB), TB, 0, stream>>>(dt_proj_w, w_dt,
            DEPTH, D_INNER, DT_RANK, D_INNER, DT_KP);
        n = DEPTH * DIM * D_INNER;
        pack_w_kernel<<<nblk(n, TB), TB, 0, stream>>>(out_proj_w, w_out,
            DEPTH, DIM, D_INNER, DIM, D_INNER);
        n = 3 * DIM * DIM;
        pack_w_kernel<<<nblk(n, TB), TB, 0, stream>>>(qkv_w, w_qkv,
            1, 3 * DIM, DIM, 3 * DIM, DIM);
        n = DIM * DIM;
        pack_w_kernel<<<nblk(n, TB), TB, 0, stream>>>(proj_w, w_proj,
            1, DIM, DIM, DIM, DIM);
    }

    // NCHW -> fp16 tokens
    {
        int n = BB * DIM * LTOK;
        reshape_in_kernel<<<nblk(n, TB), TB, 0, stream>>>(x, t16);
    }

    for (int i = 0; i < DEPTH; ++i) {
        // xz = t @ in_proj^T   (4608 x 768 x 192) -> fp32
        launch_gemm<0, true, false>(t16, DIM, w_in + (size_t)i * 2 * D_INNER * DIM, DIM,
                                    nullptr, xz, nullptr, 2 * D_INNER,
                                    NROW, 2 * D_INNER, DIM, stream);
        // depthwise causal conv + silu -> xc16
        {
            int n = NROW * D_INNER;
            conv_silu_kernel<<<nblk(n, TB), TB, 0, stream>>>(xz, conv_w, conv_b, xc16, i);
        }
        // dbc = xc @ x_proj^T  (4608 x 64pad x 384) -> fp32 + fp16
        launch_gemm<0, true, true>(xc16, D_INNER, w_x + (size_t)i * DBC_LD * D_INNER, D_INNER,
                                   nullptr, dbc32, dbc16, DBC_LD,
                                   NROW, DBC_LD, D_INNER, stream);
        // dt = softplus(dbc[:, :12] @ dt_proj^T + b)  (4608 x 384 x 32pad) -> fp32
        launch_gemm<1, true, false>(dbc16, DBC_LD, w_dt + (size_t)i * D_INNER * DT_KP, DT_KP,
                                    dt_proj_b + (size_t)i * D_INNER, dt32, nullptr, D_INNER,
                                    NROW, D_INNER, DT_KP, stream);
        // selective scan -> u16 = (y + D*xc) * silu(z)
        {
            int n = BB * D_INNER;
            scan_kernel<<<nblk(n, TB), TB, 0, stream>>>(dt32, dbc32, xc16, xz,
                                                        A_log, D_skip, u16, i);
        }
        // t = u @ out_proj^T   (4608 x 192 x 384) -> fp32 + fp16
        launch_gemm<0, true, true>(u16, D_INNER, w_out + (size_t)i * DIM * D_INNER, D_INNER,
                                   nullptr, t32, t16, DIM,
                                   NROW, DIM, D_INNER, stream);
    }

    // RMSNorm: fp32 t -> fp16 t16
    rmsnorm_kernel<<<nblk(NROW, TB), TB, 0, stream>>>(t32, norm_w, t16);

    // qkv = t @ qkv_w^T + qkv_b   (4608 x 576 x 192) -> fp32
    launch_gemm<0, true, false>(t16, DIM, w_qkv, DIM, qkv_b, qkvb, nullptr, 3 * DIM,
                                NROW, 3 * DIM, DIM, stream);

    // neighborhood attention -> attn16
    {
        int n = BB * LTOK * HEADS;
        na2d_kernel<<<nblk(n, TB), TB, 0, stream>>>(qkvb, rpb, attn16);
    }

    // proj: tn = attn @ proj_w^T + proj_b   (4608 x 192 x 192) -> fp32
    launch_gemm<0, true, false>(attn16, DIM, w_proj, DIM, proj_b, tn, nullptr, DIM,
                                NROW, DIM, DIM, stream);

    // tokens -> NCHW
    {
        int n = BB * DIM * LTOK;
        reshape_out_kernel<<<nblk(n, TB), TB, 0, stream>>>(tn, out);
    }
}
